// SASREC_69114613727866
// MI455X (gfx1250) — matrix operand order repaired and checked by reading, not compiled, not run on hardware
//
#include <hip/hip_runtime.h>
#include <hip/hip_bf16.h>
#include <math.h>

#define NUM_LAYERS 2
#define BATCH 16
#define SEQ   1024
#define DIM   128
#define HEADS 4
#define DH    32
#define MROWS (BATCH * SEQ)          /* 16384 */
#define NEGV  (-4294967295.0f)       /* -2^32 + 1, as in reference */

typedef __attribute__((ext_vector_type(16))) _Float16 v16h;
typedef __attribute__((ext_vector_type(8)))  _Float16 v8h;
typedef __attribute__((ext_vector_type(8)))  float    v8f;
typedef __attribute__((ext_vector_type(4)))  float    v4f;

// ---------------------------------------------------------------------------
// WMMA fragment loaders (layouts per CDNA5 ISA 7.12.2, wave32)
// ---------------------------------------------------------------------------

// A-matrix 16x32 f16: lane l holds row m=l&15; elements 0..7 = k hi*8+0..7,
// elements 8..15 = k 16+hi*8+0..7  (hi = l>>4).  Two contiguous 16B loads.
static __device__ __forceinline__ v16h load_a_frag(const _Float16* Ab, int lda,
                                                   int row0, int k0, int lane) {
  int m = lane & 15, hi = lane >> 4;
  const _Float16* p = Ab + (size_t)(row0 + m) * lda + k0;
  v8h lo = *(const v8h*)(p + hi * 8);
  v8h hh = *(const v8h*)(p + 16 + hi * 8);
  return __builtin_shufflevector(lo, hh, 0,1,2,3,4,5,6,7,8,9,10,11,12,13,14,15);
}

// B-matrix 32x16 f16 with contraction contiguous in memory:
// B[k][nn] = S[(n0+nn)*ld + k0 + k].  Lane l owns col n=l&15; elements 0..7 =
// k hi*8+0..7, elements 8..15 = k 16+hi*8+0..7.  Two contiguous 16B loads.
static __device__ __forceinline__ v16h load_bN_frag(const _Float16* S, int ld,
                                                    int n0, int k0, int lane) {
  int n = lane & 15, hi = lane >> 4;
  const _Float16* p = S + (size_t)(n0 + n) * ld + k0;
  v8h lo = *(const v8h*)(p + hi * 8);
  v8h hh = *(const v8h*)(p + 16 + hi * 8);
  return __builtin_shufflevector(lo, hh, 0,1,2,3,4,5,6,7,8,9,10,11,12,13,14,15);
}

static __device__ __forceinline__ v8f wmma_f16(v16h a, v16h b, v8f c) {
  return __builtin_amdgcn_wmma_f32_16x16x32_f16(false, a, false, b,
                                                (short)0, c, false, false);
}

#define VZERO ((v8f){0.f,0.f,0.f,0.f,0.f,0.f,0.f,0.f})

// ---------------------------------------------------------------------------
// T-orientation wave GEMM: computes out^T = Wt(A) @ X^T(B) for 32 rows of X.
// C frag c[s][n0]: rows m = out-col n0*16 + r + 8hi, col n = out-row.
// Lane ends up owning 8 CONSECUTIVE output columns of one row -> v8h stores.
// ---------------------------------------------------------------------------
static __device__ __forceinline__ void wave_gemmT32(const _Float16* X, int row0,
                                                    const _Float16* Wt,
                                                    v8f c[2][8], int lane) {
#pragma unroll
  for (int k0 = 0; k0 < DIM; k0 += 32) {
    v16h b0 = load_bN_frag(X, DIM, row0,      k0, lane);
    v16h b1 = load_bN_frag(X, DIM, row0 + 16, k0, lane);
#pragma unroll
    for (int n0 = 0; n0 < 8; ++n0) {
      v16h a = load_a_frag(Wt, DIM, n0 * 16, k0, lane);
      c[0][n0] = wmma_f16(a, b0, c[0][n0]);
      c[1][n0] = wmma_f16(a, b1, c[1][n0]);
    }
  }
}

// Old orientation (A = X rows, B = Wt cols): used for V whose consumer wants
// the TRANSPOSED layout -> contiguous v8h stores along the sequence dim.
static __device__ __forceinline__ void wave_gemm32(const _Float16* X, int r0,
                                                   const _Float16* Wt,
                                                   v8f c[2][8], int lane) {
#pragma unroll
  for (int k0 = 0; k0 < DIM; k0 += 32) {
    v16h a0 = load_a_frag(X, DIM, r0,      k0, lane);
    v16h a1 = load_a_frag(X, DIM, r0 + 16, k0, lane);
#pragma unroll
    for (int n0 = 0; n0 < 8; ++n0) {
      v16h bf = load_bN_frag(Wt, DIM, n0 * 16, k0, lane);
      c[0][n0] = wmma_f16(a0, bf, c[0][n0]);
      c[1][n0] = wmma_f16(a1, bf, c[1][n0]);
    }
  }
}

// Row-major f16 store for T-orientation results (+bias, optional ReLU).
template <bool RELU>
static __device__ __forceinline__ void storeT_f16(v8f c[2][8], _Float16* out,
                                                  const float* bias, int row0,
                                                  int lane) {
  int n = lane & 15, hi = lane >> 4;
#pragma unroll
  for (int n0 = 0; n0 < 8; ++n0) {
    v4f b4a = *(const v4f*)(bias + n0 * 16 + 8 * hi);
    v4f b4b = *(const v4f*)(bias + n0 * 16 + 8 * hi + 4);
#pragma unroll
    for (int s = 0; s < 2; ++s) {
      int row = row0 + s * 16 + n;
      v8h pack;
#pragma unroll
      for (int r = 0; r < 8; ++r) {
        float v = c[s][n0][r] + (r < 4 ? b4a[r] : b4b[r - 4]);
        if (RELU) v = fmaxf(v, 0.0f);
        pack[r] = (_Float16)v;
      }
      *(v8h*)(out + (size_t)row * DIM + n0 * 16 + 8 * hi) = pack;
    }
  }
}

// ---------------------------------------------------------------------------
// K0: convert all layer weights to f16, TRANSPOSED to N-major (Wt[n][k])
// ---------------------------------------------------------------------------
__global__ void convw_kernel(const float* __restrict__ Wq, const float* __restrict__ Wk,
                             const float* __restrict__ Wv, const float* __restrict__ W1,
                             const float* __restrict__ W2, _Float16* __restrict__ w16t) {
  int t = blockIdx.x * blockDim.x + threadIdx.x;
  if (t >= NUM_LAYERS * 5 * DIM * DIM) return;
  int layer = t / (5 * DIM * DIM);
  int rem   = t - layer * 5 * DIM * DIM;
  int which = rem / (DIM * DIM);
  int e     = rem - which * (DIM * DIM);        // e = k*128 + n in source
  int k = e >> 7, n = e & (DIM - 1);
  const float* src = (which == 0) ? Wq : (which == 1) ? Wk
                   : (which == 2) ? Wv : (which == 3) ? W1 : W2;
  w16t[(size_t)(layer * 5 + which) * DIM * DIM + (size_t)n * DIM + k] =
      (_Float16)src[(size_t)layer * DIM * DIM + e];
}

// ---------------------------------------------------------------------------
// K1: fused LayerNorm (per-position gamma/beta) + x->f16 + km/qm sign masks
// ---------------------------------------------------------------------------
__global__ __launch_bounds__(DIM) void ln_kernel(
    const float* __restrict__ x, const float* __restrict__ g,
    const float* __restrict__ bt, _Float16* __restrict__ x16,
    float* __restrict__ qin32, _Float16* __restrict__ qin16,
    float* __restrict__ km, float* __restrict__ qm) {
  int row = blockIdx.x;            // b*SEQ + l
  int l   = row & (SEQ - 1);
  int d   = threadIdx.x;
  __shared__ float red[DIM];

  float xv = x[(size_t)row * DIM + d];
  x16[(size_t)row * DIM + d] = (_Float16)xv;

  red[d] = xv; __syncthreads();
  for (int s = DIM / 2; s > 0; s >>= 1) { if (d < s) red[d] += red[d + s]; __syncthreads(); }
  float sum = red[0];
  __syncthreads();

  float mean = sum * (1.0f / DIM);
  float dv = xv - mean;
  red[d] = dv * dv; __syncthreads();
  for (int s = DIM / 2; s > 0; s >>= 1) { if (d < s) red[d] += red[d + s]; __syncthreads(); }
  float var = red[0] * (1.0f / DIM);
  __syncthreads();

  float qv = g[(size_t)l * DIM + d] * (dv * rsqrtf(var + 1e-8f)) + bt[(size_t)l * DIM + d];
  qin32[(size_t)row * DIM + d] = qv;
  qin16[(size_t)row * DIM + d] = (_Float16)qv;

  red[d] = qv; __syncthreads();
  for (int s = DIM / 2; s > 0; s >>= 1) { if (d < s) red[d] += red[d + s]; __syncthreads(); }
  if (d == 0) {
    km[row] = (sum    != 0.0f) ? 1.0f : 0.0f;
    qm[row] = (red[0] != 0.0f) ? 1.0f : 0.0f;
  }
}

// ---------------------------------------------------------------------------
// K2: Q/K/V projections (blockIdx.y selects which). One wave -> 32x128 tile.
// Q,K: T-orientation, row-major vector stores.  V: old orientation, stored
// TRANSPOSED (vt[(b*128+col)*SEQ + l]) with contiguous v8h stores.
// ---------------------------------------------------------------------------
__global__ __launch_bounds__(32) void qkv_kernel(
    const _Float16* __restrict__ qin16, const _Float16* __restrict__ x16,
    const _Float16* __restrict__ wbase, const float* __restrict__ bq,
    const float* __restrict__ bk, const float* __restrict__ bv,
    _Float16* __restrict__ q16, _Float16* __restrict__ k16,
    _Float16* __restrict__ vt16) {
  int r0 = blockIdx.x * 32;
  int which = blockIdx.y;
  int lane = threadIdx.x, n = lane & 15, hi = lane >> 4;

  v8f c[2][8];
#pragma unroll
  for (int s = 0; s < 2; ++s)
#pragma unroll
    for (int t = 0; t < 8; ++t) c[s][t] = VZERO;

  if (which == 2) {
    const _Float16* Wt = wbase + (size_t)2 * DIM * DIM;
    wave_gemm32(x16, r0, Wt, c, lane);
    int bidx = r0 >> 10, lbase = r0 & (SEQ - 1);
#pragma unroll
    for (int n0 = 0; n0 < 8; ++n0) {
      int col = n0 * 16 + n;
      float bval = bv[col];
#pragma unroll
      for (int s = 0; s < 2; ++s) {
        v8h pack;
#pragma unroll
        for (int r = 0; r < 8; ++r) pack[r] = (_Float16)(c[s][n0][r] + bval);
        *(v8h*)(vt16 + ((size_t)bidx * DIM + col) * SEQ + lbase + s * 16 + 8 * hi) = pack;
      }
    }
  } else {
    const _Float16* X  = (which == 0) ? qin16 : x16;
    const _Float16* Wt = wbase + (size_t)which * DIM * DIM;
    wave_gemmT32(X, r0, Wt, c, lane);
    storeT_f16<false>(c, (which == 0) ? q16 : k16, (which == 0) ? bq : bk, r0, lane);
  }
}

// ---------------------------------------------------------------------------
// K3: causal flash attention, one wave per (16 queries, head, batch).
// Computes S^T = K @ Q^T so that (a) softmax over keys is an in-lane
// reduction + one shfl_xor(16), and (b) the C-layout of S^T IS the A-layout
// of P for the P@V WMMA -> no LDS round-trip at all.
// ---------------------------------------------------------------------------
__global__ __launch_bounds__(32) void attn_kernel(
    const _Float16* __restrict__ Q, const _Float16* __restrict__ K,
    const _Float16* __restrict__ Vt, const float* __restrict__ km,
    const float* __restrict__ qm, const float* __restrict__ qin,
    float* __restrict__ ao32, _Float16* __restrict__ ao16) {
  const int i0 = blockIdx.x * 16;
  const int h  = blockIdx.y;
  const int bb = blockIdx.z;
  const int lane = threadIdx.x, n = lane & 15, hi = lane >> 4;
  const size_t rowbase = (size_t)bb * SEQ;
  const _Float16* Kb  = K  + rowbase * DIM;
  const _Float16* Vtb = Vt + (size_t)bb * DIM * SEQ;   // [col][l]

  // Q^T B-fragment: queries i0..i0+15, contraction = head dims (row-major Q)
  v16h bq = load_bN_frag(Q + rowbase * DIM, DIM, i0, h * DH, lane);

  v8f o0 = VZERO, o1 = VZERO;          // O: rows = queries, cols = dh
  float mrun = -INFINITY, lrun = 0.f;  // per-lane state for query i0 + n
  const int   iq = i0 + n;
  const float iscale = 0.17677669529663687f;   // 1/sqrt(dh)

  // j0 multiple of 32, j0 <= i0+15 <= 1023  =>  keys j0+31 <= 1023: no OOB.
  for (int j0 = 0; j0 <= i0 + 15; j0 += 32) {
    if (j0 + 32 <= i0 + 15)   // prefetch next key tile row for this lane
      __builtin_prefetch(Kb + (size_t)(j0 + 32 + n) * DIM + h * DH, 0, 0);

    // ---- S^T = K_tile @ Q^T : rows = keys (in VGPRs), cols = queries ----
    v16h ak0 = load_a_frag(Kb, DIM, j0,      h * DH, lane);
    v16h ak1 = load_a_frag(Kb, DIM, j0 + 16, h * DH, lane);
    v8f st0 = wmma_f16(ak0, bq, VZERO);   // keys j0   + 8*hi + r
    v8f st1 = wmma_f16(ak1, bq, VZERO);   // keys j0+16+ 8*hi + r

    // key-padding mask, vectorized (keys contiguous per lane)
    v4f kma = *(const v4f*)(km + rowbase + j0 + 8 * hi);
    v4f kmb = *(const v4f*)(km + rowbase + j0 + 8 * hi + 4);
    v4f kmc = *(const v4f*)(km + rowbase + j0 + 16 + 8 * hi);
    v4f kmd = *(const v4f*)(km + rowbase + j0 + 16 + 8 * hi + 4);

    float s0v[8], s1v[8];
#pragma unroll
    for (int r = 0; r < 8; ++r) {
      int ja = j0 + 8 * hi + r, jb = ja + 16;
      float ka = (r < 4) ? kma[r] : kmb[r - 4];
      float kb = (r < 4) ? kmc[r] : kmd[r - 4];
      float sA = st0[r] * iscale; if (ja > iq || ka == 0.f) sA = NEGV;
      float sB = st1[r] * iscale; if (jb > iq || kb == 0.f) sB = NEGV;
      s0v[r] = sA; s1v[r] = sB;
    }

    // ---- online softmax for query iq: in-lane reduce + one shfl_xor ----
    float tmax = fmaxf(s0v[0], s1v[0]);
#pragma unroll
    for (int r = 1; r < 8; ++r) tmax = fmaxf(tmax, fmaxf(s0v[r], s1v[r]));
    tmax = fmaxf(tmax, __shfl_xor(tmax, 16, 32));
    float mnew = fmaxf(mrun, tmax);
    float corr = __expf(mrun - mnew);

    v16h ap;                      // A-layout of P == C-layout of S^T (!)
    float rs = 0.f;
#pragma unroll
    for (int r = 0; r < 8; ++r) {
      float e0 = __expf(s0v[r] - mnew);
      float e1 = __expf(s1v[r] - mnew);
      rs += e0 + e1;
      ap[r]     = (_Float16)e0;
      ap[8 + r] = (_Float16)e1;
    }
    rs += __shfl_xor(rs, 16, 32);
    lrun = lrun * corr + rs;
    mrun = mnew;

    // rescale O (rows = queries, per-row corr gathered from lane-domain)
#pragma unroll
    for (int r = 0; r < 8; ++r) {
      float cr = __shfl(corr, r + 8 * hi, 32);
      o0[r] *= cr; o1[r] *= cr;
    }

    // ---- O += P @ V_tile  (Vt is col-major over l: contiguous k loads) ----
    v16h bv0 = load_bN_frag(Vtb, SEQ, h * DH,      j0, lane);
    v16h bv1 = load_bN_frag(Vtb, SEQ, h * DH + 16, j0, lane);
    o0 = wmma_f16(ap, bv0, o0);
    o1 = wmma_f16(ap, bv1, o1);
  }

  // epilogue: normalize, apply query mask AFTER softmax, residual on q_in
  float invn = qm[rowbase + iq] / lrun;        // per-lane (query iq)
#pragma unroll
  for (int r = 0; r < 8; ++r) {
    float inv = __shfl(invn, r + 8 * hi, 32);  // per O-row
    size_t row = rowbase + i0 + r + 8 * hi;
    size_t base = row * DIM + h * DH;
    float v0 = o0[r] * inv + qin[base + n];
    float v1 = o1[r] * inv + qin[base + 16 + n];
    ao32[base + n]      = v0;
    ao32[base + 16 + n] = v1;
    ao16[base + n]      = (_Float16)v0;
    ao16[base + 16 + n] = (_Float16)v1;
  }
}

// ---------------------------------------------------------------------------
// K4: FFN layer 1 (ReLU), K5: FFN layer 2 (+residual, *mask) — T-orientation
// ---------------------------------------------------------------------------
__global__ __launch_bounds__(32) void ffn1_kernel(
    const _Float16* __restrict__ ao16, const _Float16* __restrict__ Wt,
    const float* __restrict__ bias, _Float16* __restrict__ h16) {
  int r0 = blockIdx.x * 32;
  int lane = threadIdx.x;
  v8f c[2][8];
#pragma unroll
  for (int s = 0; s < 2; ++s)
#pragma unroll
    for (int t = 0; t < 8; ++t) c[s][t] = VZERO;
  wave_gemmT32(ao16, r0, Wt, c, lane);
  storeT_f16<true>(c, h16, bias, r0, lane);
}

__global__ __launch_bounds__(32) void ffn2_kernel(
    const _Float16* __restrict__ h16, const _Float16* __restrict__ Wt,
    const float* __restrict__ bias, const float* __restrict__ ao32,
    const float* __restrict__ maskp, float* __restrict__ xout) {
  int r0 = blockIdx.x * 32;
  int lane = threadIdx.x, n = lane & 15, hi = lane >> 4;
  v8f c[2][8];
#pragma unroll
  for (int s = 0; s < 2; ++s)
#pragma unroll
    for (int t = 0; t < 8; ++t) c[s][t] = VZERO;
  wave_gemmT32(h16, r0, Wt, c, lane);

#pragma unroll
  for (int n0 = 0; n0 < 8; ++n0) {
    v4f b4a = *(const v4f*)(bias + n0 * 16 + 8 * hi);
    v4f b4b = *(const v4f*)(bias + n0 * 16 + 8 * hi + 4);
#pragma unroll
    for (int s = 0; s < 2; ++s) {
      int row = r0 + s * 16 + n;
      float mk = maskp[row];
      const v4f* res = (const v4f*)(ao32 + (size_t)row * DIM + n0 * 16 + 8 * hi);
      v4f ra = res[0], rb = res[1];
      v4f oa, ob;
#pragma unroll
      for (int r = 0; r < 4; ++r) oa[r] = (ra[r] + c[s][n0][r]     + b4a[r]) * mk;
#pragma unroll
      for (int r = 0; r < 4; ++r) ob[r] = (rb[r] + c[s][n0][r + 4] + b4b[r]) * mk;
      v4f* dst = (v4f*)(xout + (size_t)row * DIM + n0 * 16 + 8 * hi);
      dst[0] = oa; dst[1] = ob;
    }
  }
}

// ---------------------------------------------------------------------------
// Host side
// ---------------------------------------------------------------------------
extern "C" void kernel_launch(void* const* d_in, const int* in_sizes, int n_in,
                              void* d_out, int out_size, void* d_ws, size_t ws_size,
                              hipStream_t stream) {
  const float* x_in  = (const float*)d_in[0];
  const float* maskp = (const float*)d_in[1];
  const float* g_all = (const float*)d_in[2];
  const float* b_all = (const float*)d_in[3];
  const float* Wq = (const float*)d_in[4];  const float* bq = (const float*)d_in[5];
  const float* Wk = (const float*)d_in[6];  const float* bk = (const float*)d_in[7];
  const float* Wv = (const float*)d_in[8];  const float* bv = (const float*)d_in[9];
  const float* W1 = (const float*)d_in[10]; const float* b1 = (const float*)d_in[11];
  const float* W2 = (const float*)d_in[12]; const float* b2 = (const float*)d_in[13];

  char* ws = (char*)d_ws;
  size_t off = 0;
  auto carve = [&](size_t bytes) { char* p = ws + off; off += (bytes + 255) & ~(size_t)255; return p; };
  _Float16* x16   = (_Float16*)carve((size_t)MROWS * DIM * 2);
  float*    qin32 = (float*)   carve((size_t)MROWS * DIM * 4);
  _Float16* qin16 = (_Float16*)carve((size_t)MROWS * DIM * 2);
  _Float16* q16   = (_Float16*)carve((size_t)MROWS * DIM * 2);
  _Float16* k16   = (_Float16*)carve((size_t)MROWS * DIM * 2);
  _Float16* vt16  = (_Float16*)carve((size_t)MROWS * DIM * 2);
  float*    km    = (float*)   carve((size_t)MROWS * 4);
  float*    qm    = (float*)   carve((size_t)MROWS * 4);
  float*    ao32  = (float*)   carve((size_t)MROWS * DIM * 4);
  _Float16* ao16  = (_Float16*)carve((size_t)MROWS * DIM * 2);
  _Float16* h16   = (_Float16*)carve((size_t)MROWS * DIM * 2);
  float*    xbuf  = (float*)   carve((size_t)MROWS * DIM * 4);
  _Float16* w16t  = (_Float16*)carve((size_t)NUM_LAYERS * 5 * DIM * DIM * 2);

  // K0: weights -> f16, transposed to N-major (all layers at once)
  {
    int total = NUM_LAYERS * 5 * DIM * DIM;
    convw_kernel<<<(total + 255) / 256, 256, 0, stream>>>(Wq, Wk, Wv, W1, W2, w16t);
  }

  for (int layer = 0; layer < NUM_LAYERS; ++layer) {
    const float* xcur = (layer == 0) ? x_in : xbuf;
    const _Float16* wl = w16t + (size_t)layer * 5 * DIM * DIM;

    ln_kernel<<<MROWS, DIM, 0, stream>>>(
        xcur, g_all + (size_t)layer * SEQ * DIM, b_all + (size_t)layer * SEQ * DIM,
        x16, qin32, qin16, km, qm);

    qkv_kernel<<<dim3(MROWS / 32, 3), 32, 0, stream>>>(
        qin16, x16, wl, bq + layer * DIM, bk + layer * DIM, bv + layer * DIM,
        q16, k16, vt16);

    attn_kernel<<<dim3(SEQ / 16, HEADS, BATCH), 32, 0, stream>>>(
        q16, k16, vt16, km, qm, qin32, ao32, ao16);

    ffn1_kernel<<<MROWS / 32, 32, 0, stream>>>(
        ao16, wl + 3 * DIM * DIM, b1 + layer * DIM, h16);

    float* xout = (layer == NUM_LAYERS - 1) ? (float*)d_out : xbuf;
    ffn2_kernel<<<MROWS / 32, 32, 0, stream>>>(
        h16, wl + 4 * DIM * DIM, b2 + layer * DIM, ao32, maskp, xout);
  }
}
